// SupervisedGaussKernel2d_17781164605470
// MI455X (gfx1250) — compile-verified, matrix operand order checked by reading, not run on hardware
//
#include <hip/hip_runtime.h>

typedef __attribute__((ext_vector_type(16))) _Float16 v16h;
typedef __attribute__((ext_vector_type(8)))  float    v8f;
typedef unsigned int u32x4 __attribute__((ext_vector_type(4)));
typedef int          i32x4 __attribute__((ext_vector_type(4)));
typedef int          i32x8 __attribute__((ext_vector_type(8)));

#define CHN   64
#define IMH   128
#define IMW   128
#define TILE  16
#define NCOL  18      // halo cols x0-1 .. x0+16

#if __has_builtin(__builtin_amdgcn_tensor_load_to_lds) && __has_builtin(__builtin_amdgcn_s_wait_tensorcnt)
#define HAVE_TDM 1
#else
#define HAVE_TDM 0
#endif

__device__ __forceinline__ unsigned lds_byte_off(const void* p) {
    // generic pointer to LDS: low 32 bits are the LDS byte offset (aperture in [63:32])
    return (unsigned)(unsigned long long)p;
}

#if HAVE_TDM
// 3D tile load (x,y,z) = (tile0,tile1,tile2) of a [CHN][IMH][IMW] fp32 tensor into LDS.
// TDM writes x-innermost, then y, then z -> LDS becomes channel-major [z][y][x] contiguous.
__device__ __forceinline__ void tdm_load_3d(unsigned ldsOff, const void* gaddr,
                                            int tile0, int tile1, int tile2)
{
    const unsigned long long ga = (unsigned long long)gaddr;
    u32x4 g0;
    g0[0] = 1u;                                                   // count=1, user mode
    g0[1] = ldsOff;                                               // lds_addr
    g0[2] = (unsigned)(ga & 0xFFFFFFFFull);                       // global_addr[31:0]
    g0[3] = (unsigned)((ga >> 32) & 0x01FFFFFFull) | (2u << 30);  // addr[56:32] | type=2

    i32x8 g1;
    g1[0] = (2 << 16);                                   // data_size=4B; mask/pad/iterate=0
    g1[1] = (IMW & 0xFFFF) << 16;                        // tensor_dim0[15:0] @ [63:48]
    g1[2] = ((IMW >> 16) & 0xFFFF) | ((IMH & 0xFFFF) << 16); // dim0 hi | dim1 lo
    g1[3] = ((IMH >> 16) & 0xFFFF) | (tile0 << 16);      // dim1 hi | tile_dim0
    g1[4] = (tile1 & 0xFFFF) | (tile2 << 16);            // tile_dim1 | tile_dim2
    g1[5] = IMW;                                         // tensor_dim0_stride lo (row stride)
    g1[6] = ((IMH * IMW) & 0xFFFF) << 16;                // dim0_stride hi=0 | dim1_stride lo
    g1[7] = (IMH * IMW) >> 16;                           // dim1_stride hi (channel stride)

    i32x4 g2;
    g2[0] = CHN;   // tensor_dim2
    g2[1] = 0;     // tensor_dim3 (unused)
    g2[2] = 0;     // tensor_dim2_stride lo (unused for 3D tile)
    g2[3] = 0;     // tensor_dim2_stride hi | tile_dim3=0
    i32x4 g3 = {0, 0, 0, 0};
    i32x8 g4 = {0, 0, 0, 0, 0, 0, 0, 0};   // unused trailing group (VADDR4 slot), zero-filled

    __builtin_amdgcn_tensor_load_to_lds(g0, g1, g2, g3, g4, 0);
}
#endif

// out[c,p] = f1[c,p] * sum_{q in 3x3(p) & in-image} exp(-0.5*||f1(:,p)-f2(:,q)||^2)
__global__ __launch_bounds__(32)
void pixel_adaptive_gauss_kernel(const float* __restrict__ f1,
                                 const float* __restrict__ f2,
                                 float* __restrict__ out)
{
    const int lane = threadIdx.x;           // 0..31 (wave32)
    const int x0   = blockIdx.x * TILE;
    const int y    = blockIdx.y;
    const int b    = blockIdx.z;

    // channel-major LDS (matches TDM write order); lane-stride-1 -> bank-conflict free
    __shared__ __align__(16) float s_f2[CHN * 3 * NCOL];  // [c][r][col]
    __shared__ __align__(16) float s_f1[CHN * TILE];      // [c][i]
    __shared__ float s_G[6 * 256];                        // Gram tiles [dy][tile][m][n]
    __shared__ float s_n2[3 * NCOL];
    __shared__ float s_n1[TILE];
    __shared__ float s_S[TILE];

    const size_t batch_off = (size_t)b * CHN * IMH * IMW;

    const bool interior = (y >= 1) && (y <= IMH - 2) && (x0 != 0) && (x0 != IMW - TILE);

#if HAVE_TDM
    if (interior) {
        // async DMA: f2 halo tile 18x3x64, f1 center tile 16x1x64 (all in-bounds)
        tdm_load_3d(lds_byte_off(s_f2),
                    f2 + batch_off + (size_t)(y - 1) * IMW + (x0 - 1), NCOL, 3, CHN);
        tdm_load_3d(lds_byte_off(s_f1),
                    f1 + batch_off + (size_t)y * IMW + x0, TILE, 1, CHN);
        __builtin_amdgcn_s_wait_tensorcnt(0);
    } else
#endif
    {
        // predicated fallback staging (borders / no-TDM toolchain), same LDS layout
        for (int idx = lane; idx < CHN * 3 * NCOL; idx += 32) {
            const int c   = idx / (3 * NCOL);
            const int rem = idx % (3 * NCOL);
            const int r   = rem / NCOL;
            const int col = rem % NCOL;
            const int gy  = y - 1 + r;
            const int gx  = x0 - 1 + col;
            float v = 0.0f;
            if ((unsigned)gy < IMH && (unsigned)gx < IMW)
                v = f2[batch_off + ((size_t)c * IMH + gy) * IMW + gx];
            s_f2[idx] = v;   // == c*54 + r*18 + col
        }
        for (int idx = lane; idx < CHN * TILE; idx += 32) {
            const int c = idx >> 4;
            const int i = idx & 15;
            s_f1[idx] = f1[batch_off + ((size_t)c * IMH + y) * IMW + x0 + i];
        }
    }
    // speculative prefetch of next tile's f1 (global_prefetch_b8)
    if (x0 + TILE < IMW)
        __builtin_prefetch(&f1[batch_off + (size_t)y * IMW + x0 + TILE], 0, 0);

    __syncthreads();

    // ---- exact fp32 squared norms ----
    for (int p = lane; p < 3 * NCOL; p += 32) {
        float s = 0.f;
        for (int c = 0; c < CHN; ++c) {
            const float v = s_f2[c * (3 * NCOL) + p];
            s += v * v;
        }
        s_n2[p] = s;
    }
    if (lane < TILE) {
        float s = 0.f;
        for (int c = 0; c < CHN; ++c) {
            const float v = s_f1[c * TILE + lane];
            s += v * v;
        }
        s_n1[lane] = s;
    }
    __syncthreads();

    // ---- WMMA Gram tiles: G[dy][tile][m][n] = f1(x0+m) . f2(y-1+dy, x0-1+16*tile+n)
    // 16-bit A 16x32 layout: row = lane&15; lane half selects K 0-7/16-23 vs 8-15/24-31.
    const int m  = lane & 15;
    const int kb = (lane >> 4) * 8;

    v16h afrag[2];
    for (int kc = 0; kc < 2; ++kc) {
        for (int t = 0; t < 8; ++t) {
            afrag[kc][t]     = (_Float16)s_f1[(kc * 32 + kb + t) * TILE + m];
            afrag[kc][t + 8] = (_Float16)s_f1[(kc * 32 + kb + 16 + t) * TILE + m];
        }
    }

    for (int dy = 0; dy < 3; ++dy) {
        for (int tile = 0; tile < 2; ++tile) {
            int col = tile * 16 + (lane & 15);
            if (col >= NCOL) col = NCOL - 1;    // clamp; clamped lanes' results unused
            v8f acc = {};
            for (int kc = 0; kc < 2; ++kc) {    // accumulate over K = 64 channels
                v16h bfrag;
                for (int t = 0; t < 8; ++t) {
                    bfrag[t]     = (_Float16)s_f2[(kc * 32 + kb + t) * (3 * NCOL) + dy * NCOL + col];
                    bfrag[t + 8] = (_Float16)s_f2[(kc * 32 + kb + 16 + t) * (3 * NCOL) + dy * NCOL + col];
                }
                acc = __builtin_amdgcn_wmma_f32_16x16x32_f16(
                        false, afrag[kc], false, bfrag,
                        (short)0, acc, false, false);
            }
            // D layout: lanes 0-15 -> M=v,N=lane ; lanes 16-31 -> M=8+v,N=lane-16
            const int Mbase = (lane >> 4) * 8;
            const int N     = lane & 15;
            float* g = &s_G[(dy * 2 + tile) * 256];
            for (int v = 0; v < 8; ++v)
                g[(Mbase + v) * 16 + N] = acc[v];
        }
    }
    __syncthreads();

    // ---- per-pixel adaptive weight S(i) over valid 3x3 neighbors ----
    if (lane < TILE) {
        const int   i  = lane;
        const float ni = s_n1[i];
        float S = 0.f;
        for (int dy = 0; dy < 3; ++dy) {
            const int gy = y - 1 + dy;
            if ((unsigned)gy >= IMH) continue;
            for (int dx = 0; dx < 3; ++dx) {
                const int gx = x0 + i + dx - 1;
                if ((unsigned)gx >= IMW) continue;
                const int jj = i + dx;          // 0..17 column into halo strip
                const int tl = jj >> 4, n = jj & 15;
                const float cross = s_G[(dy * 2 + tl) * 256 + i * 16 + n];
                const float d2 = ni + s_n2[dy * NCOL + jj] - 2.0f * cross;
                S += __expf(-0.5f * d2);
            }
        }
        s_S[i] = S;
    }
    __syncthreads();

    // ---- out[c, y, x0+i] = f1[c, y, x0+i] * S[i]  (fp32 exact) ----
    {
        const int   i = lane & 15;
        const int   g = lane >> 4;
        const float S = s_S[i];
        for (int t = 0; t < CHN / 2; ++t) {
            const int c = g * (CHN / 2) + t;
            out[batch_off + ((size_t)c * IMH + y) * IMW + x0 + i] =
                s_f1[c * TILE + i] * S;
        }
    }
}

extern "C" void kernel_launch(void* const* d_in, const int* in_sizes, int n_in,
                              void* d_out, int out_size, void* d_ws, size_t ws_size,
                              hipStream_t stream) {
    const float* f1 = (const float*)d_in[0];
    const float* f2 = (const float*)d_in[1];
    float* out = (float*)d_out;

    const int bs = in_sizes[0] / (CHN * IMH * IMW);   // 4 for the reference shapes

    dim3 grid(IMW / TILE, IMH, bs);
    dim3 block(32);
    pixel_adaptive_gauss_kernel<<<grid, block, 0, stream>>>(f1, f2, out);
}
